// MultiHeadAttention_47115791237226
// MI455X (gfx1250) — matrix operand order repaired and checked by reading, not compiled, not run on hardware
//
#include <hip/hip_runtime.h>
#include <hip/hip_bf16.h>

// ---------------------------------------------------------------------------
// MI455X (gfx1250) fused MHA:
//   Phase 1: Q/K/V = X @ W{q,k,v}^T   -- bf16 WMMA, f32 accum, bf16 out to ws
//            128x256x32 block tile, 8 waves, 64x64 per wave (4x4 WMMA tiles),
//            double-buffered LDS, running-pointer global prefetch.
//   Phase 2: per-token 16x16 head-attention (S = Q K^T via WMMA, masked
//            softmax in C-layout regs, O = P V via coalesced VALU)
// Workspace requirement: 3 * 16384 * 2048 * 2 B = 192 MiB (bf16 Q,K,V).
// ---------------------------------------------------------------------------

typedef __attribute__((ext_vector_type(16))) __bf16 v16bf;
typedef __attribute__((ext_vector_type(8)))  __bf16 v8bf;
typedef __attribute__((ext_vector_type(4)))  __bf16 v4bf;
typedef __attribute__((ext_vector_type(8)))  float  v8f;

#define DIMS     2048
#define HEADS    16
#define HEAD_DIM 128
#define M_TOK    16384          // B*L = 4*4096

#define BM 128
#define BN 256
#define BK 32
#define KT_STEPS (DIMS / BK)    // 64
#define PITCH 40                // LDS row pitch (bf16): 32 + 8 pad, 16B-aligned rows,
                                // stride 20 words -> conflict-free b128 fragment reads

static __device__ __forceinline__ v16bf cat8(v8bf lo, v8bf hi) {
  return __builtin_shufflevector(lo, hi, 0,1,2,3,4,5,6,7,8,9,10,11,12,13,14,15);
}

static __device__ __forceinline__ v4bf cvt4(float4 f) {
  v4bf c;
  c[0] = (__bf16)f.x; c[1] = (__bf16)f.y; c[2] = (__bf16)f.z; c[3] = (__bf16)f.w;
  return c;
}

// ---------------------------------------------------------------------------
// Phase 1: Y[m,n] = sum_k X[m,k] * W[n,k]   (both K-contiguous: "NT" GEMM)
// ---------------------------------------------------------------------------
__global__ __launch_bounds__(256) void qkv_proj_kernel(
    const float* __restrict__ X,
    const float* __restrict__ Wq,
    const float* __restrict__ Wk,
    const float* __restrict__ Wv,
    __bf16* __restrict__ qkv)
{
  __shared__ __bf16 As[2][BM * PITCH];   // 2 * 128*40*2 = 20480 B
  __shared__ __bf16 Bs[2][BN * PITCH];   // 2 * 256*40*2 = 40960 B

  const float* __restrict__ W =
      (blockIdx.z == 0) ? Wq : (blockIdx.z == 1) ? Wk : Wv;
  __bf16* __restrict__ Y = qkv + (size_t)blockIdx.z * (size_t)M_TOK * DIMS;

  const int t    = threadIdx.x;
  const int lane = t & 31;
  const int wid  = t >> 5;             // 0..7
  const int wmo  = (wid >> 2) * 64;    // wave row offset: 0 / 64
  const int wno  = (wid & 3)  * 64;    // wave col offset: 0 / 64 / 128 / 192
  const int li   = lane & 15;
  const int hlf  = lane >> 4;          // 0 or 1

  const int bm0 = blockIdx.y * BM;
  const int bn0 = blockIdx.x * BN;

  // Staging map: thread t covers rows (t/8 + 32*i) with k-chunk (t%8)*4.
  // Row offsets i*32*DIMS*4 B (= i*256 KB) fold into the 24-bit load IOFFSET.
  const int srow = t >> 3;             // 0..31
  const int sc4  = (t & 7) * 4;        // k offset within tile

  const float* xp = X + (size_t)(bm0 + srow) * DIMS + sc4;   // running pointer
  const float* wp = W + (size_t)(bn0 + srow) * DIMS + sc4;   // running pointer

  v8f acc[4][4];
  #pragma unroll
  for (int mt = 0; mt < 4; ++mt)
    #pragma unroll
    for (int nt = 0; nt < 4; ++nt)
      acc[mt][nt] = (v8f){0.f,0.f,0.f,0.f,0.f,0.f,0.f,0.f};

  float4 ra[4], rb[8];

  // -- prologue: tile 0 -> regs -> LDS buf0; tile 1 -> regs (in flight)
  #pragma unroll
  for (int i = 0; i < 4; ++i) ra[i] = *(const float4*)(xp + (size_t)i * 32 * DIMS);
  #pragma unroll
  for (int i = 0; i < 8; ++i) rb[i] = *(const float4*)(wp + (size_t)i * 32 * DIMS);
  xp += BK; wp += BK;

  #pragma unroll
  for (int i = 0; i < 4; ++i) *(v4bf*)&As[0][(srow + 32*i) * PITCH + sc4] = cvt4(ra[i]);
  #pragma unroll
  for (int i = 0; i < 8; ++i) *(v4bf*)&Bs[0][(srow + 32*i) * PITCH + sc4] = cvt4(rb[i]);

  #pragma unroll
  for (int i = 0; i < 4; ++i) ra[i] = *(const float4*)(xp + (size_t)i * 32 * DIMS);
  #pragma unroll
  for (int i = 0; i < 8; ++i) rb[i] = *(const float4*)(wp + (size_t)i * 32 * DIMS);
  xp += BK; wp += BK;

  for (int kt = 0; kt < KT_STEPS; ++kt) {
    const int cur = kt & 1;
    const int nxt = cur ^ 1;
    __syncthreads();   // buf[cur] fully written; buf[nxt] readers (kt-1) done

    // -- compute on buf[cur]: 8 fragment loads + 16 WMMAs
    v16bf af[4], bfr[4];
    #pragma unroll
    for (int mt = 0; mt < 4; ++mt) {
      const __bf16* base = &As[cur][(wmo + mt*16 + li) * PITCH + hlf*8];
      af[mt] = cat8(*(const v8bf*)base, *(const v8bf*)(base + 16));
    }
    #pragma unroll
    for (int nt = 0; nt < 4; ++nt) {
      const __bf16* base = &Bs[cur][(wno + nt*16 + li) * PITCH + hlf*8];
      bfr[nt] = cat8(*(const v8bf*)base, *(const v8bf*)(base + 16));
    }
    #pragma unroll
    for (int mt = 0; mt < 4; ++mt)
      #pragma unroll
      for (int nt = 0; nt < 4; ++nt)
        acc[mt][nt] = __builtin_amdgcn_wmma_f32_16x16x32_bf16(
            false, af[mt], false, bfr[nt], (short)0, acc[mt][nt], false, false);

    // -- stage tile kt+1 (already in regs) into buf[nxt]
    if (kt + 1 < KT_STEPS) {
      #pragma unroll
      for (int i = 0; i < 4; ++i) *(v4bf*)&As[nxt][(srow + 32*i) * PITCH + sc4] = cvt4(ra[i]);
      #pragma unroll
      for (int i = 0; i < 8; ++i) *(v4bf*)&Bs[nxt][(srow + 32*i) * PITCH + sc4] = cvt4(rb[i]);
    }
    // -- issue global loads for tile kt+2 (running pointers, imm row offsets)
    if (kt + 2 < KT_STEPS) {
      #pragma unroll
      for (int i = 0; i < 4; ++i) ra[i] = *(const float4*)(xp + (size_t)i * 32 * DIMS);
      #pragma unroll
      for (int i = 0; i < 8; ++i) rb[i] = *(const float4*)(wp + (size_t)i * 32 * DIMS);
      xp += BK; wp += BK;
    }
  }

  // epilogue: C layout -> bf16 global (lane li = col, VGPR r = row r + hlf*8)
  #pragma unroll
  for (int mt = 0; mt < 4; ++mt)
    #pragma unroll
    for (int nt = 0; nt < 4; ++nt)
      #pragma unroll
      for (int r = 0; r < 8; ++r) {
        const int row = bm0 + wmo + mt*16 + hlf*8 + r;
        const int col = bn0 + wno + nt*16 + li;
        Y[(size_t)row * DIMS + col] = (__bf16)acc[mt][nt][r];
      }
}

// ---------------------------------------------------------------------------
// Phase 2: one wave per token. HEADS=16 => S = Q K^T is one 16x16 WMMA chain
// (Dh=128 => 4x wmma 16x16x32 bf16). Masked softmax in C-layout registers
// with 16-lane shfl_xor row reductions, then O = P @ V with coalesced loads.
// ---------------------------------------------------------------------------
__global__ __launch_bounds__(256) void attn_kernel(
    const __bf16* __restrict__ qkv, float* __restrict__ out)
{
  __shared__ float pLds[8][16][17];   // per-wave 16x16 P, pitch 17 (bank-safe)

  const int lane  = threadIdx.x & 31;
  const int wv    = threadIdx.x >> 5;
  const int li    = lane & 15;
  const int hlf   = lane >> 4;
  const int token = blockIdx.x * 8 + wv;

  const __bf16* __restrict__ Qp = qkv + (size_t)token * DIMS;
  const __bf16* __restrict__ Kp = qkv + (size_t)1 * M_TOK * DIMS + (size_t)token * DIMS;
  const __bf16* __restrict__ Vp = qkv + (size_t)2 * M_TOK * DIMS + (size_t)token * DIMS;

  // S = Q K^T : A/B fragments read directly from global in WMMA lane layout
  v8f s = (v8f){0.f,0.f,0.f,0.f,0.f,0.f,0.f,0.f};
  #pragma unroll
  for (int kk = 0; kk < HEAD_DIM; kk += 32) {
    const __bf16* qb = Qp + li * HEAD_DIM + kk + hlf*8;
    v16bf a = cat8(*(const v8bf*)qb, *(const v8bf*)(qb + 16));
    const __bf16* kb = Kp + li * HEAD_DIM + kk + hlf*8;
    v16bf b = cat8(*(const v8bf*)kb, *(const v8bf*)(kb + 16));
    s = __builtin_amdgcn_wmma_f32_16x16x32_bf16(false, a, false, b, (short)0, s, false, false);
  }

  // causal mask over (h,g) = (row m, col n=li), softmax over n. Row m lives in
  // lanes {hlf*16 .. hlf*16+15}; shfl_xor masks 1,2,4,8 stay inside the half.
  #pragma unroll
  for (int r = 0; r < 8; ++r) {
    const int m = hlf*8 + r;
    const bool valid = (li <= m);
    float x = valid ? s[r] : -__builtin_inff();
    float mx = x;
    mx = fmaxf(mx, __shfl_xor(mx, 1));
    mx = fmaxf(mx, __shfl_xor(mx, 2));
    mx = fmaxf(mx, __shfl_xor(mx, 4));
    mx = fmaxf(mx, __shfl_xor(mx, 8));
    float e = valid ? __expf(s[r] - mx) : 0.f;
    float sum = e;
    sum += __shfl_xor(sum, 1);
    sum += __shfl_xor(sum, 2);
    sum += __shfl_xor(sum, 4);
    sum += __shfl_xor(sum, 8);
    pLds[wv][m][li] = e / sum;
  }
  // same-wave LDS RAW fence (split counters: DS ops tracked by DScnt)
  asm volatile("s_wait_dscnt 0" ::: "memory");

  // O[m,d] = sum_g P[m,g] * V[g,d]; lane owns d = lane + 32*j, j=0..3
  float acc[16][4];
  #pragma unroll
  for (int m = 0; m < 16; ++m)
    #pragma unroll
    for (int j = 0; j < 4; ++j) acc[m][j] = 0.f;

  #pragma unroll
  for (int g = 0; g < 16; ++g) {
    float vvl[4];
    #pragma unroll
    for (int j = 0; j < 4; ++j)
      vvl[j] = (float)Vp[g * HEAD_DIM + lane + 32*j];
    #pragma unroll
    for (int m = 0; m < 16; ++m) {
      const float pm = pLds[wv][m][g];   // LDS broadcast read
      #pragma unroll
      for (int j = 0; j < 4; ++j) acc[m][j] += pm * vvl[j];
    }
  }

  float* __restrict__ Op = out + (size_t)token * DIMS;
  #pragma unroll
  for (int m = 0; m < 16; ++m)
    #pragma unroll
    for (int j = 0; j < 4; ++j)
      Op[m * HEAD_DIM + lane + 32*j] = acc[m][j];
}

// ---------------------------------------------------------------------------
extern "C" void kernel_launch(void* const* d_in, const int* in_sizes, int n_in,
                              void* d_out, int out_size, void* d_ws, size_t ws_size,
                              hipStream_t stream) {
  const float* X  = (const float*)d_in[0];
  const float* Wq = (const float*)d_in[1];
  const float* Wk = (const float*)d_in[2];
  const float* Wv = (const float*)d_in[3];
  float* out      = (float*)d_out;
  __bf16* qkv_ws  = (__bf16*)d_ws;   // needs 192 MiB (bf16 Q,K,V)

  dim3 g1(DIMS / BN, M_TOK / BM, 3);      // (8, 128, 3)
  qkv_proj_kernel<<<g1, 256, 0, stream>>>(X, Wq, Wk, Wv, qkv_ws);

  attn_kernel<<<M_TOK / 8, 256, 0, stream>>>(qkv_ws, out);
}
